// MiniMaxM2SparseMoeBlock_66571993088731
// MI455X (gfx1250) — compile-verified, matrix operand order checked by reading, not run on hardware
//
#include <hip/hip_runtime.h>
#include <hip/hip_bf16.h>
#include <math.h>

// Problem constants (from reference: B=4, S=4096, H=2048, I=1024, E=8, top-2)
#define T_NUM 16384
#define H_DIM 2048
#define I_DIM 1024
#define E_NUM 8

// LDS tile geometry: 128 N-rows x 64 K elems (bf16), padded to 72 elems/row.
#define KCHUNK   64
#define LDS_PITCH 72

typedef __attribute__((ext_vector_type(16))) __bf16 v16bf;
typedef __attribute__((ext_vector_type(8)))  __bf16 v8bf;
typedef __attribute__((ext_vector_type(4)))  __bf16 v4bf;
typedef __attribute__((ext_vector_type(8)))  float  v8f;

// ---------------------------------------------------------------------------
// Raw 32-bit LDS offset of a __shared__ object (addrspace(3) ptrtoint),
// as required by the VDST field of global_load_async_to_lds_*.
// ---------------------------------------------------------------------------
__device__ __forceinline__ unsigned lds_off(const void* p) {
  return (unsigned)(uintptr_t)(const __attribute__((address_space(3))) void*)p;
}

// One per-lane 16-byte async global->LDS copy (ASYNCcnt-tracked).
__device__ __forceinline__ void async_copy16(unsigned lds_addr, const void* g) {
  unsigned long long ga = (unsigned long long)(uintptr_t)g;
  asm volatile("global_load_async_to_lds_b128 %0, %1, off"
               :: "v"(lds_addr), "v"(ga) : "memory");
}

// ---------------------------------------------------------------------------
// Build one 16x32 bf16 WMMA A/B fragment from an fp32 row (x-matrix A
// fragment in the up/gate GEMM; amortized over 8 WMMAs).
// Layout per ISA: lane holds K = k0+8*half+[0..7] in v0..3 and
// K = k0+16+8*half+[0..7] in v4..7 (half = lane>>4).
// ---------------------------------------------------------------------------
__device__ __forceinline__ v16bf make_frag_f32(const float* __restrict__ row,
                                               int k0, int half) {
  const float4* p0 = reinterpret_cast<const float4*>(row + k0 + 8 * half);
  const float4* p1 = reinterpret_cast<const float4*>(row + k0 + 16 + 8 * half);
  float4 a0 = p0[0], a1 = p0[1];
  float4 b0 = p1[0], b1 = p1[1];
  v16bf f;
  f[0]  = (__bf16)a0.x; f[1]  = (__bf16)a0.y; f[2]  = (__bf16)a0.z; f[3]  = (__bf16)a0.w;
  f[4]  = (__bf16)a1.x; f[5]  = (__bf16)a1.y; f[6]  = (__bf16)a1.z; f[7]  = (__bf16)a1.w;
  f[8]  = (__bf16)b0.x; f[9]  = (__bf16)b0.y; f[10] = (__bf16)b0.z; f[11] = (__bf16)b0.w;
  f[12] = (__bf16)b1.x; f[13] = (__bf16)b1.y; f[14] = (__bf16)b1.z; f[15] = (__bf16)b1.w;
  return f;
}

// Fragment from a bf16 row in global or LDS: two 16-byte loads.
__device__ __forceinline__ v16bf make_frag_bf16(const __bf16* row,
                                                int k0, int half) {
  v8bf lo = *reinterpret_cast<const v8bf*>(row + k0 + 8 * half);
  v8bf hi = *reinterpret_cast<const v8bf*>(row + k0 + 16 + 8 * half);
  v16bf f;
#pragma unroll
  for (int i = 0; i < 8; ++i) { f[i] = lo[i]; f[8 + i] = hi[i]; }
  return f;
}

// Cooperative async fill of one [128 x KCHUNK] bf16 tile into LDS.
// 256 threads x 4 iters x 16B = 16 KB. 4 async ops per thread.
__device__ __forceinline__ void fill_tile_async(const __bf16* __restrict__ gbase,
                                                int n0, int kc, int g_pitch,
                                                __bf16* sdst) {
  const int tid = threadIdx.x;
#pragma unroll
  for (int i = 0; i < 4; ++i) {
    int idx = tid + i * 256;          // 0..1023 16B segments
    int row = idx >> 3;               // 0..127
    int seg = idx & 7;                // 8 x 16B = 128B per row
    const __bf16* g = gbase + (size_t)(n0 + row) * g_pitch + kc + seg * 8;
    async_copy16(lds_off(sdst + row * LDS_PITCH + seg * 8), g);
  }
}

// ---------------------------------------------------------------------------
// Zero-fill helper (graph-capture-safe replacement for hipMemsetAsync).
// ---------------------------------------------------------------------------
__global__ void zero_f32_kernel(float* __restrict__ p, size_t n) {
  size_t i = (size_t)blockIdx.x * blockDim.x + threadIdx.x;
  if (i < n) p[i] = 0.0f;
}

// ---------------------------------------------------------------------------
// fp32 -> bf16 bulk conversion (per-expert weight staging; memory-bound).
// ---------------------------------------------------------------------------
__global__ __launch_bounds__(256)
void cvt_f32_bf16_kernel(const float* __restrict__ src,
                         __bf16* __restrict__ dst, size_t n4) {
  size_t i = (size_t)blockIdx.x * blockDim.x + threadIdx.x;
  if (i < n4) {
    float4 v = reinterpret_cast<const float4*>(src)[i];
    v4bf o;
    o[0] = (__bf16)v.x; o[1] = (__bf16)v.y;
    o[2] = (__bf16)v.z; o[3] = (__bf16)v.w;
    reinterpret_cast<v4bf*>(dst)[i] = o;
  }
}

// ---------------------------------------------------------------------------
// Router: logits = x @ gate_w.T, sigmoid scores, biased top-2, normalized
// routing weights; compacts (token, weight) pairs into per-expert lists.
// ---------------------------------------------------------------------------
__global__ __launch_bounds__(256)
void router_kernel(const float* __restrict__ x,
                   const float* __restrict__ gw,
                   const float* __restrict__ ebias,
                   float* __restrict__ logits_out,   // [T, E]
                   int* __restrict__ cnt,            // [E]
                   int* __restrict__ idxbuf,         // [E, T]
                   float* __restrict__ wbuf) {       // [E, T]
  const int wave = threadIdx.x >> 5;
  const int lane = threadIdx.x & 31;
  const int t = blockIdx.x * 8 + wave;
  const float* xrow = x + (size_t)t * H_DIM;

  float acc[E_NUM];
#pragma unroll
  for (int e = 0; e < E_NUM; ++e) acc[e] = 0.0f;

  for (int h = lane * 4; h < H_DIM; h += 32 * 4) {
    float4 xv = *reinterpret_cast<const float4*>(xrow + h);
#pragma unroll
    for (int e = 0; e < E_NUM; ++e) {
      float4 gv = *reinterpret_cast<const float4*>(gw + (size_t)e * H_DIM + h);
      acc[e] += xv.x * gv.x + xv.y * gv.y + xv.z * gv.z + xv.w * gv.w;
    }
  }
#pragma unroll
  for (int e = 0; e < E_NUM; ++e) {
#pragma unroll
    for (int off = 16; off > 0; off >>= 1)
      acc[e] += __shfl_xor(acc[e], off, 32);
  }

  if (lane == 0) {
    float score[E_NUM], biased[E_NUM];
#pragma unroll
    for (int e = 0; e < E_NUM; ++e) {
      float lg = acc[e];
      logits_out[(size_t)t * E_NUM + e] = lg;
      float s = 1.0f / (1.0f + __expf(-lg));
      score[e]  = s;
      biased[e] = s + ebias[e];
    }
    int s0 = 0;
#pragma unroll
    for (int e = 1; e < E_NUM; ++e) if (biased[e] > biased[s0]) s0 = e;
    int s1 = (s0 == 0) ? 1 : 0;
#pragma unroll
    for (int e = 0; e < E_NUM; ++e)
      if (e != s0 && biased[e] > biased[s1]) s1 = e;

    float sum = fmaxf(score[s0] + score[s1], 1e-12f);
    float w0 = score[s0] / sum;
    float w1 = score[s1] / sum;

    int p0 = atomicAdd(&cnt[s0], 1);
    idxbuf[(size_t)s0 * T_NUM + p0] = t;
    wbuf[(size_t)s0 * T_NUM + p0]  = w0;
    int p1 = atomicAdd(&cnt[s1], 1);
    idxbuf[(size_t)s1 * T_NUM + p1] = t;
    wbuf[(size_t)s1 * T_NUM + p1]  = w1;
  }
}

// ---------------------------------------------------------------------------
// Per-expert up/gate GEMM: h[r, i] = silu(x[tok_r] . w1[i]) * (x[tok_r] . w3[i])
// Block tile 64(M) x 128(N=I). w1/w3 K-chunks are double-buffered in LDS via
// ASYNCcnt-tracked global_load_async_to_lds_b128; WMMAs consume ds_load_b128
// fragments while the async engine fills the next chunk.
// ---------------------------------------------------------------------------
__global__ __launch_bounds__(256)
void moe_up_gate_kernel(const float* __restrict__ x,
                        const __bf16* __restrict__ w1b,  // [I, H] bf16
                        const __bf16* __restrict__ w3b,  // [I, H] bf16
                        const int* __restrict__ idx,     // [T] compacted tokens
                        const int* __restrict__ cntp,
                        __bf16* __restrict__ hbuf) {     // [T, I]
  __shared__ __bf16 sb[2][2][128 * LDS_PITCH];   // 72 KB

  const int cnt = *cntp;
  const int m_blk = blockIdx.x * 64;
  if (m_blk >= cnt) return;                      // block-uniform: no barrier yet

  const int wave   = threadIdx.x >> 5;
  const int lane   = threadIdx.x & 31;
  const int lhalf  = lane >> 4;
  const int l15    = lane & 15;
  const int m_sub  = wave & 3;
  const int n_half = wave >> 2;
  const int m_tile = m_blk + m_sub * 16;
  const int n_blk  = blockIdx.y * 128;

  const int m  = m_tile + l15;
  const int mc = (m < cnt) ? m : (cnt - 1);
  const int tok = idx[mc];
  const float* xrow = x + (size_t)tok * H_DIM;

  v8f acc1[4] = {{}, {}, {}, {}};
  v8f acc3[4] = {{}, {}, {}, {}};

  // Prologue: fill buffer 0 (8 async ops / thread).
  fill_tile_async(w1b, n_blk, 0, H_DIM, sb[0][0]);
  fill_tile_async(w3b, n_blk, 0, H_DIM, sb[0][1]);

  int buf = 0;
  for (int kc = 0; kc < H_DIM; kc += KCHUNK) {
    const int nb = buf ^ 1;
    if (kc + KCHUNK < H_DIM) {
      fill_tile_async(w1b, n_blk, kc + KCHUNK, H_DIM, sb[nb][0]);
      fill_tile_async(w3b, n_blk, kc + KCHUNK, H_DIM, sb[nb][1]);
      asm volatile("s_wait_asynccnt 8" ::: "memory");  // current buf done
    } else {
      asm volatile("s_wait_asynccnt 0" ::: "memory");
    }
    __syncthreads();

#pragma unroll
    for (int ks = 0; ks < 2; ++ks) {
      v16bf afrag = make_frag_f32(xrow, kc + ks * 32, lhalf);
#pragma unroll
      for (int j = 0; j < 4; ++j) {
        const int nloc = n_half * 64 + j * 16 + l15;
        const __bf16* s1 = &sb[buf][0][nloc * LDS_PITCH];
        const __bf16* s3 = &sb[buf][1][nloc * LDS_PITCH];
        v16bf b1 = make_frag_bf16(s1, ks * 32, lhalf);
        v16bf b3 = make_frag_bf16(s3, ks * 32, lhalf);
        acc1[j] = __builtin_amdgcn_wmma_f32_16x16x32_bf16(
            false, afrag, false, b1, (short)0, acc1[j], false, false);
        acc3[j] = __builtin_amdgcn_wmma_f32_16x16x32_bf16(
            false, afrag, false, b3, (short)0, acc3[j], false, false);
      }
    }
    __syncthreads();    // all reads of buf done before it is overwritten
    buf = nb;
  }

  // Epilogue: h = silu(c1) * c3, store bf16. C layout: n = lane&15,
  // m = v + 8*(lane>>4) within the 16x16 tile.
#pragma unroll
  for (int j = 0; j < 4; ++j) {
    const int n = n_blk + n_half * 64 + j * 16 + l15;
#pragma unroll
    for (int v = 0; v < 8; ++v) {
      const int mr = m_tile + v + 8 * lhalf;
      if (mr < cnt) {
        float g = acc1[j][v];
        float u = acc3[j][v];
        float hv = (g / (1.0f + __expf(-g))) * u;   // silu(g) * up
        hbuf[(size_t)mr * I_DIM + n] = (__bf16)hv;
      }
    }
  }
}

// ---------------------------------------------------------------------------
// Per-expert down GEMM + weighted scatter:
//   out[tok_r, n] += w_r * (h[r] . w2[n]),  K = I = 1024.
// w2 K-chunks double-buffered in LDS via async copies; h rows (bf16) direct.
// Experts serialized on the stream -> race-free deterministic RMW.
// ---------------------------------------------------------------------------
__global__ __launch_bounds__(256)
void moe_down_kernel(const __bf16* __restrict__ hbuf,   // [T, I]
                     const __bf16* __restrict__ w2b,    // [H, I] bf16
                     const int* __restrict__ idx,
                     const float* __restrict__ wbuf,
                     const int* __restrict__ cntp,
                     float* __restrict__ out) {         // [T, H]
  __shared__ __bf16 sb[2][128 * LDS_PITCH];   // 36 KB

  const int cnt = *cntp;
  const int m_blk = blockIdx.x * 64;
  if (m_blk >= cnt) return;

  const int wave   = threadIdx.x >> 5;
  const int lane   = threadIdx.x & 31;
  const int lhalf  = lane >> 4;
  const int l15    = lane & 15;
  const int m_sub  = wave & 3;
  const int n_half = wave >> 2;
  const int m_tile = m_blk + m_sub * 16;
  const int n_blk  = blockIdx.y * 128;

  const int m  = m_tile + l15;
  const int mc = (m < cnt) ? m : (cnt - 1);
  const __bf16* hrow = hbuf + (size_t)mc * I_DIM;

  v8f acc[4] = {{}, {}, {}, {}};

  fill_tile_async(w2b, n_blk, 0, I_DIM, sb[0]);

  int buf = 0;
  for (int kc = 0; kc < I_DIM; kc += KCHUNK) {
    const int nb = buf ^ 1;
    if (kc + KCHUNK < I_DIM) {
      fill_tile_async(w2b, n_blk, kc + KCHUNK, I_DIM, sb[nb]);
      asm volatile("s_wait_asynccnt 4" ::: "memory");
    } else {
      asm volatile("s_wait_asynccnt 0" ::: "memory");
    }
    __syncthreads();

#pragma unroll
    for (int ks = 0; ks < 2; ++ks) {
      v16bf afrag = make_frag_bf16(hrow, kc + ks * 32, lhalf);
#pragma unroll
      for (int j = 0; j < 4; ++j) {
        const int nloc = n_half * 64 + j * 16 + l15;
        v16bf b2 = make_frag_bf16(&sb[buf][nloc * LDS_PITCH], ks * 32, lhalf);
        acc[j] = __builtin_amdgcn_wmma_f32_16x16x32_bf16(
            false, afrag, false, b2, (short)0, acc[j], false, false);
      }
    }
    __syncthreads();
    buf = nb;
  }

#pragma unroll
  for (int j = 0; j < 4; ++j) {
    const int n = n_blk + n_half * 64 + j * 16 + l15;
#pragma unroll
    for (int v = 0; v < 8; ++v) {
      const int mr = m_tile + v + 8 * lhalf;
      if (mr < cnt) {
        const int   t   = idx[mr];
        const float wgt = wbuf[mr];
        float* o = out + (size_t)t * H_DIM + n;
        *o += wgt * acc[j][v];
      }
    }
  }
}

// ---------------------------------------------------------------------------
// Host-side orchestration. Inputs (setup_inputs order):
//   0: hidden_states [B,S,H] f32   1: gate_w [E,H] f32   2: e_bias [E] f32
//   3: w1 [E,I,H] f32              4: w2 [E,H,I] f32     5: w3 [E,I,H] f32
// d_out: out [T*H] f32 then router_logits [T*E] f32.
// Workspace layout (~46 MB):
//   [0, 64)           counts (E ints)
//   [256, +512K)      idxbuf  [E,T] int
//   [.., +512K)       wbuf    [E,T] f32
//   [2 MB, +32 MB)    hbuf    [T,I] bf16
//   [34 MB, +4 MB)    w1b     [I,H] bf16 (per-expert staging)
//   [38 MB, +4 MB)    w3b     [I,H] bf16
//   [42 MB, +4 MB)    w2b     [H,I] bf16
// ---------------------------------------------------------------------------
extern "C" void kernel_launch(void* const* d_in, const int* in_sizes, int n_in,
                              void* d_out, int out_size, void* d_ws, size_t ws_size,
                              hipStream_t stream) {
  (void)in_sizes; (void)n_in; (void)out_size; (void)ws_size;

  const float* x  = (const float*)d_in[0];
  const float* gw = (const float*)d_in[1];
  const float* eb = (const float*)d_in[2];
  const float* w1 = (const float*)d_in[3];
  const float* w2 = (const float*)d_in[4];
  const float* w3 = (const float*)d_in[5];

  float* out    = (float*)d_out;
  float* logits = out + (size_t)T_NUM * H_DIM;

  char*   ws     = (char*)d_ws;
  int*    cnt    = (int*)ws;
  int*    idxbuf = (int*)(ws + 256);
  float*  wbuf   = (float*)(ws + 256 + (size_t)E_NUM * T_NUM * 4);
  __bf16* hbuf   = (__bf16*)(ws + ((size_t)2  << 20));
  __bf16* w1b    = (__bf16*)(ws + ((size_t)34 << 20));
  __bf16* w3b    = (__bf16*)(ws + ((size_t)38 << 20));
  __bf16* w2b    = (__bf16*)(ws + ((size_t)42 << 20));

  // Zero counters + output accumulator (kernel-based: graph-capture safe).
  zero_f32_kernel<<<1, 64, 0, stream>>>((float*)cnt, 16);
  {
    const size_t n = (size_t)T_NUM * H_DIM;
    zero_f32_kernel<<<(unsigned)((n + 255) / 256), 256, 0, stream>>>(out, n);
  }

  router_kernel<<<T_NUM / 8, 256, 0, stream>>>(x, gw, eb, logits,
                                               cnt, idxbuf, wbuf);

  const size_t wElems = (size_t)I_DIM * H_DIM;      // per-expert matrix elems
  const size_t n4     = wElems / 4;
  const unsigned cvtGrid = (unsigned)((n4 + 255) / 256);

  dim3 gridA(T_NUM / 64, I_DIM / 128);   // 256 x 8
  dim3 gridB(T_NUM / 64, H_DIM / 128);   // 256 x 16
  for (int e = 0; e < E_NUM; ++e) {
    // Stage this expert's weights as bf16 (one-time per launch, L2-resident).
    cvt_f32_bf16_kernel<<<cvtGrid, 256, 0, stream>>>(w1 + (size_t)e * wElems, w1b, n4);
    cvt_f32_bf16_kernel<<<cvtGrid, 256, 0, stream>>>(w3 + (size_t)e * wElems, w3b, n4);
    cvt_f32_bf16_kernel<<<cvtGrid, 256, 0, stream>>>(w2 + (size_t)e * wElems, w2b, n4);

    moe_up_gate_kernel<<<gridA, 256, 0, stream>>>(
        x, w1b, w3b,
        idxbuf + (size_t)e * T_NUM,
        cnt + e,
        hbuf);
    moe_down_kernel<<<gridB, 256, 0, stream>>>(
        hbuf, w2b,
        idxbuf + (size_t)e * T_NUM,
        wbuf + (size_t)e * T_NUM,
        cnt + e,
        out);
  }
}
